// EncoderDecoderLSTM_32882269618875
// MI455X (gfx1250) — matrix operand order repaired and checked by reading, not compiled, not run on hardware
//
#include <hip/hip_runtime.h>

// ---------------------------------------------------------------------------
// EncoderDecoder LSTM for MI455X (gfx1250, wave32, WMMA).
// Strategy:
//   * prep kernels: convert Whh (enc+dec) fp32 -> bf16 (L2-resident, 32MB each),
//     combine biases, zero h ping-pong buffers / barrier state / out row 0.
//   * one persistent cooperative kernel (32 WGs x 256 thr) runs all timesteps.
//     Each WG owns 64 hidden units: 4 slices x 16 units; each slice computed by
//     a pair of waves splitting K=2048 in halves; gate sums via
//     v_wmma_f32_16x16x32_bf16 using the "broadcast-A" matvec trick:
//        A[m,k] = h[k]  (all 16 rows identical)
//        B[k,n] = Whh[row_n, k]   (transposed 16-row weight tile)
//     -> D[0,n] (VGPR0, lanes 0..15) = dot(Whh[row_n,:], h) partials.
//     Cell state c stays in registers of the kh==0 wave across all steps.
//   * h double-buffered in global bf16; grid-wide sense barrier per step.
//   * teacher_forcing_ratio == 1 => next decoder input is the teacher token,
//     so logits (the actual outputs) are computed by WG0/wave0 concurrently
//     with the next step (safe via double buffer); ratio < 1 falls back to a
//     two-barrier argmax-feedback path.
// ---------------------------------------------------------------------------

#define H_      2048
#define FH_     (4 * H_)      // 8192 gate rows
#define FEAT_   6
#define NC_     6
#define NWG_    32
#define BLK_    256
#define SLICES_ 4             // hidden slices (of 16) per WG
#define KHALF_  (H_ / 2)      // K split across wave pairs

typedef __attribute__((ext_vector_type(16))) __bf16 v16bf;
typedef __attribute__((ext_vector_type(8)))  __bf16 v8bf;
typedef __attribute__((ext_vector_type(8)))  float  v8f;

__device__ __forceinline__ float sigm_(float x) {
  return 1.0f / (1.0f + __expf(-x));
}
__device__ __forceinline__ float tanh_(float x) {
  return 2.0f / (1.0f + __expf(-2.0f * x)) - 1.0f;
}

// Grid-wide sense barrier (persistent kernel, NWG_ co-resident WGs).
__device__ __forceinline__ void grid_sync(unsigned* cnt, unsigned* gen) {
  __syncthreads();
  if (threadIdx.x == 0) {
    unsigned g = __hip_atomic_load(gen, __ATOMIC_RELAXED, __HIP_MEMORY_SCOPE_AGENT);
    __threadfence();  // make this WG's h-slice stores device-visible
    unsigned a = __hip_atomic_fetch_add(cnt, 1u, __ATOMIC_ACQ_REL, __HIP_MEMORY_SCOPE_AGENT);
    if (a == NWG_ - 1) {
      __hip_atomic_store(cnt, 0u, __ATOMIC_RELAXED, __HIP_MEMORY_SCOPE_AGENT);
      __hip_atomic_fetch_add(gen, 1u, __ATOMIC_ACQ_REL, __HIP_MEMORY_SCOPE_AGENT);
    } else {
      while (__hip_atomic_load(gen, __ATOMIC_ACQUIRE, __HIP_MEMORY_SCOPE_AGENT) == g) {
        __builtin_amdgcn_s_sleep(2);
      }
    }
  }
  __syncthreads();
  __threadfence();  // acquire: invalidate L0 so fresh h is visible to all lanes
}

// 4-gate WMMA accumulation over one K half.
// lh     : LDS copy of h (bf16, H_ elements)
// r0..r3 : bf16 weight row base for gate i/f/g/o (row = gate*H_ + j0 + (lane&15))
__device__ __forceinline__ void gates_wmma(const __bf16* lh,
                                           const __bf16* r0, const __bf16* r1,
                                           const __bf16* r2, const __bf16* r3,
                                           int kb, int sel8, int sel16,
                                           v8f& ai, v8f& af, v8f& ag, v8f& ao) {
#pragma unroll 2
  for (int kc = 0; kc < KHALF_; kc += 32) {
    const int k = kb + kc;
    // A fragment: h broadcast, same contiguous K order as B:
    //  low lanes: elems 0-15 = K+0..15
    //  high lanes: elems 0-15 = K+16..31.
    v8bf lo = *(const v8bf*)(lh + k + sel16);
    v8bf hi = *(const v8bf*)(lh + k + sel16 + 8);
    v16bf A = __builtin_shufflevector(lo, hi,
        0, 1, 2, 3, 4, 5, 6, 7, 8, 9, 10, 11, 12, 13, 14, 15);
    // B fragment: transposed weight tile. Lane n&15 = output row, 16 contiguous
    // K values (low lanes K+0..15, high lanes K+16..31).
    v16bf B0 = *(const v16bf*)(r0 + k + sel16);
    ai = __builtin_amdgcn_wmma_f32_16x16x32_bf16(false, A, false, B0, (short)0, ai, false, false);
    v16bf B1 = *(const v16bf*)(r1 + k + sel16);
    af = __builtin_amdgcn_wmma_f32_16x16x32_bf16(false, A, false, B1, (short)0, af, false, false);
    v16bf B2 = *(const v16bf*)(r2 + k + sel16);
    ag = __builtin_amdgcn_wmma_f32_16x16x32_bf16(false, A, false, B2, (short)0, ag, false, false);
    v16bf B3 = *(const v16bf*)(r3 + k + sel16);
    ao = __builtin_amdgcn_wmma_f32_16x16x32_bf16(false, A, false, B3, (short)0, ao, false, false);
  }
}

// ---------------------------------------------------------------------------
__global__ void prep_convert(const float* __restrict__ encWhh,
                             const float* __restrict__ decWhh,
                             __bf16* __restrict__ encW,
                             __bf16* __restrict__ decW) {
  const size_t n = (size_t)FH_ * H_;
  const size_t stride = (size_t)gridDim.x * blockDim.x;
  for (size_t i = (size_t)blockIdx.x * blockDim.x + threadIdx.x; i < n; i += stride) {
    encW[i] = (__bf16)encWhh[i];
    decW[i] = (__bf16)decWhh[i];
  }
}

__global__ void prep_misc(const float* __restrict__ ebih, const float* __restrict__ ebhh,
                          const float* __restrict__ dbih, const float* __restrict__ dbhh,
                          float* __restrict__ encB, float* __restrict__ decB,
                          __bf16* __restrict__ hb, unsigned* cnt, unsigned* gen,
                          float* predPtr, float* out) {
  const int stride = gridDim.x * blockDim.x;
  const int gid = blockIdx.x * blockDim.x + threadIdx.x;
  for (int k = gid; k < FH_; k += stride) {
    encB[k] = ebih[k] + ebhh[k];
    decB[k] = dbih[k] + dbhh[k];
  }
  for (int k = gid; k < 2 * H_; k += stride) hb[k] = (__bf16)0.0f;
  if (gid < NC_) out[gid] = 0.0f;  // outputs[0, 0, :] == 0
  if (gid == 0) { *cnt = 0u; *gen = 0u; *predPtr = 0.0f; }
}

// ---------------------------------------------------------------------------
__global__ __launch_bounds__(BLK_, 1)
void lstm_persistent(const float* __restrict__ x, const int* __restrict__ y,
                     const int* __restrict__ tfr,
                     const float* __restrict__ encWih, const float* __restrict__ decWih,
                     const float* __restrict__ linW, const float* __restrict__ linb,
                     const __bf16* __restrict__ encW, const __bf16* __restrict__ decW,
                     const float* __restrict__ encB, const float* __restrict__ decB,
                     __bf16* __restrict__ hb,   // [2][H_] ping-pong
                     unsigned* cnt, unsigned* gen, float* predPtr,
                     float* __restrict__ out, int T_enc, int L) {
  __shared__ __bf16 lds_h[H_];
  __shared__ float red[SLICES_][4][16];

  const int tid   = threadIdx.x;
  const int lane  = tid & 31;
  const int wid   = tid >> 5;
  const int slice = wid >> 1;
  const int kh    = wid & 1;          // which K half this wave reduces
  const int n16   = lane & 15;
  const int j0    = blockIdx.x * (SLICES_ * 16) + slice * 16;
  const int jrow  = j0 + n16;         // weight row within a gate block
  const int jme   = j0 + lane;        // hidden index (valid for lane < 16)
  const int sel8  = (lane >= 16) ? 8 : 0;
  const int sel16 = (lane >= 16) ? 16 : 0;
  const int kb    = kh * KHALF_;
  const bool alwaysTF = (tfr[0] >= 1);

  const __bf16* rE[4];
  const __bf16* rD[4];
#pragma unroll
  for (int g = 0; g < 4; ++g) {
    rE[g] = encW + (size_t)(g * H_ + jrow) * H_;
    rD[g] = decW + (size_t)(g * H_ + jrow) * H_;
  }

  float cst = 0.0f;  // register-resident cell state (kh==0 waves, lanes 0..15)

  // ---------------- encoder ----------------
  for (int t = 0; t < T_enc; ++t) {
    const int p = t & 1;
    const __bf16* hbR = hb + p * H_;
    __bf16* hbW = hb + (p ^ 1) * H_;

    for (int i = tid; i < H_; i += BLK_) lds_h[i] = hbR[i];
    __syncthreads();

    v8f ai = {}, af = {}, ag = {}, ao = {};
    gates_wmma(lds_h, rE[0], rE[1], rE[2], rE[3], kb, sel8, sel16, ai, af, ag, ao);

    if (kh == 1 && lane < 16) {
      red[slice][0][lane] = ai[0];
      red[slice][1][lane] = af[0];
      red[slice][2][lane] = ag[0];
      red[slice][3][lane] = ao[0];
    }
    __syncthreads();

    if (kh == 0 && lane < 16) {
      float g4[4] = { ai[0] + red[slice][0][lane] + encB[0 * H_ + jme],
                      af[0] + red[slice][1][lane] + encB[1 * H_ + jme],
                      ag[0] + red[slice][2][lane] + encB[2 * H_ + jme],
                      ao[0] + red[slice][3][lane] + encB[3 * H_ + jme] };
      const float* xt = x + (size_t)t * FEAT_;
#pragma unroll
      for (int f = 0; f < FEAT_; ++f) {
        const float xv = xt[f];
#pragma unroll
        for (int g = 0; g < 4; ++g)
          g4[g] += encWih[(size_t)(g * H_ + jme) * FEAT_ + f] * xv;
      }
      const float iv = sigm_(g4[0]), fv = sigm_(g4[1]);
      const float gv = tanh_(g4[2]), ov = sigm_(g4[3]);
      cst = fv * cst + iv * gv;
      hbW[jme] = (__bf16)(ov * tanh_(cst));
    }
    grid_sync(cnt, gen);
  }

  // ---------------- decoder ----------------
  float inp = (float)y[0];
  for (int s = 0; s < L - 1; ++s) {
    const int u = T_enc + s;
    const int p = u & 1;
    const __bf16* hbR = hb + p * H_;
    __bf16* hbW = hb + (p ^ 1) * H_;

    for (int i = tid; i < H_; i += BLK_) lds_h[i] = hbR[i];
    __syncthreads();

    v8f ai = {}, af = {}, ag = {}, ao = {};
    gates_wmma(lds_h, rD[0], rD[1], rD[2], rD[3], kb, sel8, sel16, ai, af, ag, ao);

    if (kh == 1 && lane < 16) {
      red[slice][0][lane] = ai[0];
      red[slice][1][lane] = af[0];
      red[slice][2][lane] = ag[0];
      red[slice][3][lane] = ao[0];
    }
    __syncthreads();

    if (kh == 0 && lane < 16) {
      float g4[4] = { ai[0] + red[slice][0][lane] + decB[0 * H_ + jme] + decWih[0 * H_ + jme] * inp,
                      af[0] + red[slice][1][lane] + decB[1 * H_ + jme] + decWih[1 * H_ + jme] * inp,
                      ag[0] + red[slice][2][lane] + decB[2 * H_ + jme] + decWih[2 * H_ + jme] * inp,
                      ao[0] + red[slice][3][lane] + decB[3 * H_ + jme] + decWih[3 * H_ + jme] * inp };
      const float iv = sigm_(g4[0]), fv = sigm_(g4[1]);
      const float gv = tanh_(g4[2]), ov = sigm_(g4[3]);
      cst = fv * cst + iv * gv;
      hbW[jme] = (__bf16)(ov * tanh_(cst));
    }
    grid_sync(cnt, gen);  // new h visible everywhere

    // Logits (= the outputs) from the freshly written h; WG0/wave0 only.
    // Double buffering makes this safe to overlap with the next step.
    if (blockIdx.x == 0 && wid == 0) {
      const __bf16* hn = hb + ((u + 1) & 1) * H_;
      float a0 = 0, a1 = 0, a2 = 0, a3 = 0, a4 = 0, a5 = 0;
      for (int k = lane; k < H_; k += 32) {
        const float hv = (float)hn[k];
        a0 += linW[0 * H_ + k] * hv;  a1 += linW[1 * H_ + k] * hv;
        a2 += linW[2 * H_ + k] * hv;  a3 += linW[3 * H_ + k] * hv;
        a4 += linW[4 * H_ + k] * hv;  a5 += linW[5 * H_ + k] * hv;
      }
#pragma unroll
      for (int m = 16; m; m >>= 1) {
        a0 += __shfl_xor(a0, m); a1 += __shfl_xor(a1, m); a2 += __shfl_xor(a2, m);
        a3 += __shfl_xor(a3, m); a4 += __shfl_xor(a4, m); a5 += __shfl_xor(a5, m);
      }
      if (lane == 0) {
        float lg[NC_] = { a0 + linb[0], a1 + linb[1], a2 + linb[2],
                          a3 + linb[3], a4 + linb[4], a5 + linb[5] };
        float* op = out + (size_t)(s + 1) * NC_;
        int best = 0; float bv = lg[0];
#pragma unroll
        for (int c = 0; c < NC_; ++c) {
          op[c] = lg[c];
          if (lg[c] > bv) { bv = lg[c]; best = c; }
        }
        if (!alwaysTF) *predPtr = (float)best;
      }
    }

    if (alwaysTF) {
      inp = (float)y[s + 1];          // teacher forcing (ratio >= 1 => always)
    } else {
      grid_sync(cnt, gen);            // pred visible (approx: no JAX RNG replay)
      inp = *predPtr;
    }
  }
}

// ---------------------------------------------------------------------------
extern "C" void kernel_launch(void* const* d_in, const int* in_sizes, int n_in,
                              void* d_out, int out_size, void* d_ws, size_t ws_size,
                              hipStream_t stream) {
  const float* x      = (const float*)d_in[0];
  const int*   y      = (const int*)  d_in[1];
  const int*   tfr    = (const int*)  d_in[2];
  const float* encWih = (const float*)d_in[3];
  const float* encWhh = (const float*)d_in[4];
  const float* encBih = (const float*)d_in[5];
  const float* encBhh = (const float*)d_in[6];
  const float* decWih = (const float*)d_in[7];
  const float* decWhh = (const float*)d_in[8];
  const float* decBih = (const float*)d_in[9];
  const float* decBhh = (const float*)d_in[10];
  const float* linW   = (const float*)d_in[11];
  const float* linb   = (const float*)d_in[12];
  float* out = (float*)d_out;

  const int T_enc = in_sizes[0] / FEAT_;
  const int L     = in_sizes[1];

  char* ws = (char*)d_ws;
  size_t off = 0;
  auto take = [&](size_t bytes) -> void* {
    void* p = ws + off;
    off = (off + bytes + 255) & ~(size_t)255;
    return p;
  };
  __bf16*   encW    = (__bf16*)  take((size_t)FH_ * H_ * sizeof(__bf16));
  __bf16*   decW    = (__bf16*)  take((size_t)FH_ * H_ * sizeof(__bf16));
  float*    encB    = (float*)   take((size_t)FH_ * sizeof(float));
  float*    decB    = (float*)   take((size_t)FH_ * sizeof(float));
  __bf16*   hbuf    = (__bf16*)  take((size_t)2 * H_ * sizeof(__bf16));
  unsigned* cnt     = (unsigned*)take(sizeof(unsigned));
  unsigned* gen     = (unsigned*)take(sizeof(unsigned));
  float*    predPtr = (float*)   take(sizeof(float));

  prep_convert<<<4096, 256, 0, stream>>>(encWhh, decWhh, encW, decW);
  prep_misc<<<64, 256, 0, stream>>>(encBih, encBhh, decBih, decBhh,
                                    encB, decB, hbuf, cnt, gen, predPtr, out);
  lstm_persistent<<<NWG_, BLK_, 0, stream>>>(x, y, tfr, encWih, decWih, linW, linb,
                                             encW, decW, encB, decB, hbuf,
                                             cnt, gen, predPtr, out, T_enc, L);
}
